// ScannedRNN_2233382994533
// MI455X (gfx1250) — matrix operand order repaired and checked by reading, not compiled, not run on hardware
//
#include <hip/hip_runtime.h>
#include <cstdint>
#include <cstddef>

// ---------------------------------------------------------------------------
// GRU scan for MI455X (gfx1250, wave32, WMMA).
// T=512, B=256, H=512.
//   x_proj fused per-step:  [16,512](x_t) @ Wi[512,1536]  (bf16 WMMA, f32 acc)
//   recurrence:             [16,512](h)   @ [Wh_rz|Wh_n]  (bf16 WMMA, f32 acc)
// One persistent workgroup per 16-row batch tile: no cross-WG sync needed.
// Weights pre-transposed+converted to bf16 in d_ws, laid out as SIX contiguous
// [512,512] blocks (Wi_r,Wi_z,Wi_n,Wh_r,Wh_z,Wh_n): all six B fragments for a
// column tile sit at constant <2.6MB offsets from ONE base pointer, so they
// fold into global_load_b128 24-bit immediates (2 address VGPRs per wave).
// ---------------------------------------------------------------------------

typedef __bf16 bf16;
typedef __attribute__((ext_vector_type(8)))  bf16  v8bf;
typedef __attribute__((ext_vector_type(16))) bf16  v16bf;
typedef __attribute__((ext_vector_type(8)))  float v8f;

#define T_DIM 512
#define B_DIM 256
#define H_DIM 512
#define BLK   ((size_t)H_DIM * H_DIM)   // elements per [512,512] weight block

__device__ __forceinline__ bf16 f2bf(float f) {
  union { float f; uint32_t u; } a; a.f = f;
  uint32_t r = a.u + 0x7FFFu + ((a.u >> 16) & 1u);   // round-to-nearest-even
  union { uint16_t u; bf16 b; } c; c.u = (uint16_t)(r >> 16);
  return c.b;
}

__device__ __forceinline__ v16bf cat16(v8bf lo, v8bf hi) {
  return __builtin_shufflevector(lo, hi, 0,1,2,3,4,5,6,7,8,9,10,11,12,13,14,15);
}

// Fast saturating gate math (overflow-safe: exp->inf => rcp->0).
__device__ __forceinline__ float fast_sigmoid(float x) {
  return __builtin_amdgcn_rcpf(1.0f + __expf(-x));
}
__device__ __forceinline__ float fast_tanh(float x) {
  return 1.0f - 2.0f * __builtin_amdgcn_rcpf(1.0f + __expf(2.0f * x));
}

// Transpose + f32->bf16 convert: in[K,N] row-major -> out[N,K] row-major.
// Because N is a multiple of 512 and blocks are consecutive in N, this output
// IS the desired sequence of [512,512] column blocks.
__global__ void transpose_cvt(const float* __restrict__ in, bf16* __restrict__ out,
                              int K, int N) {
  size_t i = (size_t)blockIdx.x * blockDim.x + threadIdx.x;
  if (i >= (size_t)K * N) return;
  int k = (int)(i / (size_t)N);
  int n = (int)(i % (size_t)N);
  out[(size_t)n * K + k] = f2bf(in[i]);
}

__global__ __launch_bounds__(512) void gru_scan_kernel(
    const float* __restrict__ h0,
    const float* __restrict__ ins,
    const unsigned char* __restrict__ resets,   // jnp.bool_ -> 1 byte/elem
    const float* __restrict__ bi,
    const float* __restrict__ bhn,
    const bf16* __restrict__ W,                 // 6 x [512,512] bf16 blocks
    float* __restrict__ out)                    // [T, B, H] f32
{
  const int tid    = threadIdx.x;
  const int lane   = tid & 31;
  const int wave   = tid >> 5;            // 0..15
  const int btile  = blockIdx.x;          // 0..15 : rows btile*16 .. +15
  const int n16    = lane & 15;           // WMMA N / A-row index
  const int half   = lane >> 4;           // lane half
  const int rowoff = half * 8;            // C/D rows handled by this lane

  __shared__ bf16  sh_h[16][H_DIM];       // h (bf16) for WMMA A
  __shared__ bf16  sh_x[16][H_DIM];       // x_t (bf16) for WMMA A
  __shared__ float s_bi[3 * H_DIM];
  __shared__ float s_bhn[H_DIM];

  for (int i = tid; i < 3 * H_DIM; i += 512) s_bi[i]  = bi[i];
  for (int i = tid; i < H_DIM;     i += 512) s_bhn[i] = bhn[i];

  // Two 16-column tiles of H per wave.
  const int jts[2] = { 2 * wave, 2 * wave + 1 };

  // B fragment bases. Layout (32x16 bf16 B): lanes 0-15 hold K {kb+0..7, kb+16..23} of
  // column N=lane; lanes 16-31 hold K {kb+8..15, kb+24..31} of column N=lane-16.
  // With W^T [N,K] storage that is two runs of 8 contiguous bf16 per lane; matrix m of
  // {xr,xz,xn,hr,hz,hn} is at constant element offset m*BLK (imm-foldable).
  const bf16* Wb0 = W + (size_t)(jts[0] * 16 + n16) * H_DIM + (size_t)(half * 8);
  const bf16* Wb1 = W + (size_t)(jts[1] * 16 + n16) * H_DIM + (size_t)(half * 8);

  // Carried hidden state in registers: rows rowoff..rowoff+7, col jts[q]*16+n16.
  // This exactly matches the 16x16 f32 C/D lane layout, so z*h_old is local.
  float hreg[2][8];
  #pragma unroll
  for (int q = 0; q < 2; ++q)
    #pragma unroll
    for (int i = 0; i < 8; ++i)
      hreg[q][i] = h0[(size_t)(btile * 16 + rowoff + i) * H_DIM + (jts[q] * 16 + n16)];

  // A fragment addressing (16-bit A 16x32): lanes 0-15: K {kb+0..7, kb+16..23};
  // lanes 16-31: K {kb+8..15, kb+24..31}. Row M = lane&15.
  const int aK = half * 8;
  const int am = n16;

  for (int t = 0; t < T_DIM; ++t) {
    // ------- stage: apply episode resets, publish bf16 h and x_t to LDS ----
    // 8 flags for this lane's rows, one aligned 8-byte load.
    const uint64_t rbits =
        *(const uint64_t*)(resets + (size_t)t * B_DIM + btile * 16 + rowoff);
    #pragma unroll
    for (int i = 0; i < 8; ++i) {
      if (((rbits >> (8 * i)) & 0xFFu) != 0) { hreg[0][i] = 0.0f; hreg[1][i] = 0.0f; }
    }
    #pragma unroll
    for (int q = 0; q < 2; ++q)
      #pragma unroll
      for (int i = 0; i < 8; ++i)
        sh_h[rowoff + i][jts[q] * 16 + n16] = f2bf(hreg[q][i]);

    const float* xsrc = ins + ((size_t)t * B_DIM + btile * 16) * H_DIM;
    for (int idx = tid; idx < 16 * H_DIM; idx += 512)
      sh_x[idx >> 9][idx & (H_DIM - 1)] = f2bf(xsrc[idx]);

    __syncthreads();

    // ------- GEMM phase: 16 K-steps x (2 tiles x 6 matrices) WMMA ----------
    v8f acc[2][6];
    #pragma unroll
    for (int q = 0; q < 2; ++q)
      #pragma unroll
      for (int m = 0; m < 6; ++m)
        #pragma unroll
        for (int i = 0; i < 8; ++i) acc[q][m][i] = 0.0f;

    #pragma unroll 1   // keep live ranges small: no full unroll, no spills
    for (int kb = 0; kb < H_DIM; kb += 32) {
      v16bf Ax = cat16(*(const v8bf*)&sh_x[am][kb + aK],
                       *(const v8bf*)&sh_x[am][kb + aK + 16]);
      v16bf Ah = cat16(*(const v8bf*)&sh_h[am][kb + aK],
                       *(const v8bf*)&sh_h[am][kb + aK + 16]);
      const bf16* p0 = Wb0 + kb;
      const bf16* p1 = Wb1 + kb;
      #pragma unroll
      for (int m = 0; m < 6; ++m) {
        v16bf Bv0 = cat16(*(const v8bf*)(p0 + m * BLK),
                          *(const v8bf*)(p0 + m * BLK + 16));
        acc[0][m] = __builtin_amdgcn_wmma_f32_16x16x32_bf16(
            false, (m < 3) ? Ax : Ah, false, Bv0, (short)0, acc[0][m],
            false, false);
        v16bf Bv1 = cat16(*(const v8bf*)(p1 + m * BLK),
                          *(const v8bf*)(p1 + m * BLK + 16));
        acc[1][m] = __builtin_amdgcn_wmma_f32_16x16x32_bf16(
            false, (m < 3) ? Ax : Ah, false, Bv1, (short)0, acc[1][m],
            false, false);
      }
    }

    // ------- elementwise GRU gates + output + carry update -----------------
    #pragma unroll
    for (int q = 0; q < 2; ++q) {
      const int col = jts[q] * 16 + n16;
      const float bir = s_bi[col];
      const float biz = s_bi[H_DIM + col];
      const float bin = s_bi[2 * H_DIM + col];
      const float bnn = s_bhn[col];
      #pragma unroll
      for (int i = 0; i < 8; ++i) {
        float r  = fast_sigmoid(acc[q][0][i] + bir + acc[q][3][i]);
        float z  = fast_sigmoid(acc[q][1][i] + biz + acc[q][4][i]);
        float nn = fast_tanh(acc[q][2][i] + bin + r * (acc[q][5][i] + bnn));
        float h  = (1.0f - z) * nn + z * hreg[q][i];
        hreg[q][i] = h;
        out[((size_t)t * B_DIM + btile * 16 + rowoff + i) * H_DIM + col] = h;
      }
    }
    __syncthreads();   // protect sh_h/sh_x before next stage overwrites them
  }
}

extern "C" void kernel_launch(void* const* d_in, const int* in_sizes, int n_in,
                              void* d_out, int out_size, void* d_ws, size_t ws_size,
                              hipStream_t stream) {
  (void)in_sizes; (void)n_in; (void)out_size; (void)ws_size;

  const float*         h0     = (const float*)d_in[0];
  const float*         ins    = (const float*)d_in[1];
  const unsigned char* resets = (const unsigned char*)d_in[2]; // bool_ = 1 byte
  const float*         Wi     = (const float*)d_in[3];
  const float*         bi     = (const float*)d_in[4];
  const float*         Whrz   = (const float*)d_in[5];
  const float*         Whn    = (const float*)d_in[6];
  const float*         bhn    = (const float*)d_in[7];
  float*               out    = (float*)d_out;

  // Workspace: six contiguous [512,512] bf16 blocks (3,145,728 bytes total):
  //   [0..2] = Wi^T (r,z,n)   [3..4] = Wh_rz^T (r,z)   [5] = Wh_n^T
  bf16* W = (bf16*)d_ws;
  transpose_cvt<<<(H_DIM * 3 * H_DIM + 255) / 256, 256, 0, stream>>>(Wi,   W,           H_DIM, 3 * H_DIM);
  transpose_cvt<<<(H_DIM * 2 * H_DIM + 255) / 256, 256, 0, stream>>>(Whrz, W + 3 * BLK, H_DIM, 2 * H_DIM);
  transpose_cvt<<<(H_DIM * 1 * H_DIM + 255) / 256, 256, 0, stream>>>(Whn,  W + 5 * BLK, H_DIM, 1 * H_DIM);

  gru_scan_kernel<<<dim3(B_DIM / 16), dim3(512), 0, stream>>>(
      h0, ins, resets, bi, bhn, W, out);
}
